// Attention_27685359190017
// MI455X (gfx1250) — matrix operand order repaired and checked by reading, not compiled, not run on hardware
//
#include <hip/hip_runtime.h>

#define Bb 32
#define Tt 2048
#define Dd 1024
#define Hh 1024

typedef __attribute__((ext_vector_type(16))) __bf16        v16bf;
typedef __attribute__((ext_vector_type(8)))  float         v8f;
typedef __attribute__((ext_vector_type(4)))  unsigned int  u32x4;
typedef __attribute__((ext_vector_type(2)))  unsigned int  u32x2;

union Frag { v16bf v; u32x4 q[2]; };

__device__ __forceinline__ unsigned short f2bf(float f) {
    unsigned int u = __float_as_uint(f);
    u += 0x7FFFu + ((u >> 16) & 1u);          // round-to-nearest-even
    return (unsigned short)(u >> 16);
}

// ---------------------------------------------------------------------------
// Kernel P: reorder W1[f32 D x H] into bf16 WMMA B-fragments.
// Fragment layout: [nt(64)][ks(32)][lane(32)][j(16)] u16, where
//   n = nt*16 + lane%16 ;  k = ks*32 + (lane>=16 ? 16 : 0) + j
// ---------------------------------------------------------------------------
__global__ void __launch_bounds__(256) prep_w1(const float* __restrict__ W1,
                                               unsigned short* __restrict__ w1p) {
    unsigned idx  = blockIdx.x * 256u + threadIdx.x;     // 1M elements
    unsigned j    = idx & 15u;
    unsigned lane = (idx >> 4) & 31u;
    unsigned ks   = (idx >> 9) & 31u;
    unsigned nt   = idx >> 14;
    unsigned k = ks * 32u + ((lane & 16u) ? 16u : 0u) + j;
    unsigned n = nt * 16u + (lane & 15u);
    w1p[idx] = f2bf(W1[(size_t)k * Hh + n]);
}

// ---------------------------------------------------------------------------
// Kernel H: hpb[b,h] = hidden[b,:]·W2[:,h] + b1[h] + b2[h]
// ---------------------------------------------------------------------------
__global__ void __launch_bounds__(256) hidden_proj(const float* __restrict__ hidden,
                                                   const float* __restrict__ W2,
                                                   const float* __restrict__ b1,
                                                   const float* __restrict__ b2,
                                                   float* __restrict__ hpb) {
    unsigned idx = blockIdx.x * 256u + threadIdx.x;      // 32768
    unsigned b = idx >> 10;
    unsigned h = idx & 1023u;
    const float* hid = hidden + (size_t)b * Dd;
    float acc = b1[h] + b2[h];
    #pragma unroll 4
    for (int d = 0; d < Dd; ++d)
        acc += hid[d] * W2[(size_t)d * Hh + h];
    hpb[idx] = acc;
}

// ---------------------------------------------------------------------------
// Kernel S: fused  score[b,t] = sum_h tanh( (features@W1)[t,h] + hpb[b,h] ) * V[h]
// Block = 256 thr (8 waves), 128 rows of one batch. A staged in LDS (bf16).
// ---------------------------------------------------------------------------
constexpr int AROW    = 1032;                 // 1024 + 8 bf16 pad (bank spread)
constexpr int A_BYTES = 128 * AROW * 2;       // 264192
constexpr int SMEM_S  = A_BYTES + 2 * 1024 * 4; // + hpb + V staging = 272384

__global__ void __launch_bounds__(256) score_kernel(const float* __restrict__ features,
                                                    const unsigned short* __restrict__ w1p,
                                                    const float* __restrict__ hpb,
                                                    const float* __restrict__ Vv,
                                                    float* __restrict__ scores) {
    extern __shared__ unsigned char smem[];
    unsigned short* smA = (unsigned short*)smem;
    float* smHP = (float*)(smem + A_BYTES);
    float* smV  = smHP + Hh;

    const int tid = threadIdx.x;
    const int b   = blockIdx.x >> 4;          // 16 blocks per batch
    const int t0  = (blockIdx.x & 15) << 7;   // 128 rows

    // ---- stage A: features[b, t0..t0+127, :] -> bf16 LDS ----
    {
        const float4* src = (const float4*)(features + ((size_t)b * Tt + t0) * Dd);
        for (int row = 0; row < 128; ++row) {
            float4 x = src[(size_t)row * 256 + tid];
            u32x2 p;
            p.x = (unsigned)f2bf(x.x) | ((unsigned)f2bf(x.y) << 16);
            p.y = (unsigned)f2bf(x.z) | ((unsigned)f2bf(x.w) << 16);
            *(u32x2*)(smA + row * AROW + (tid << 2)) = p;
        }
        for (int i = tid; i < Hh; i += 256) {
            smHP[i] = hpb[b * Hh + i];
            smV[i]  = Vv[i];
        }
    }
    __syncthreads();

    const int lane = tid & 31;
    const int w    = tid >> 5;                // wave id = M-tile
    const int lm   = lane & 15;
    const int hi   = lane >> 4;
    const unsigned aBase = (unsigned)(w * 16 + lm) * AROW;   // u16 units
    const u32x4* bptr = (const u32x4*)w1p;

    float partial[8];
    #pragma unroll
    for (int r = 0; r < 8; ++r) partial[r] = 0.f;

    for (int ng = 0; ng < 16; ++ng) {         // groups of 4 N-tiles
        v8f acc[4];
        #pragma unroll
        for (int i = 0; i < 4; ++i) acc[i] = v8f{0.f,0.f,0.f,0.f,0.f,0.f,0.f,0.f};
        const int ntBase = ng << 2;
        for (int ks = 0; ks < 32; ++ks) {
            Frag a;
            const int kb = (ks << 5) + (hi << 4);
            a.q[0] = *(const u32x4*)(smA + aBase + kb);
            a.q[1] = *(const u32x4*)(smA + aBase + kb + 8);
            Frag bf[4];
            size_t base = ((size_t)(ntBase * 32 + ks) * 32 + lane) * 2;  // u32x4 units
            #pragma unroll
            for (int i = 0; i < 4; ++i) {
                bf[i].q[0] = bptr[base + (size_t)i * 2048];
                bf[i].q[1] = bptr[base + (size_t)i * 2048 + 1];
            }
            #pragma unroll
            for (int i = 0; i < 4; ++i)
                acc[i] = __builtin_amdgcn_wmma_f32_16x16x32_bf16(
                    false, a.v, false, bf[i].v, (short)0, acc[i], false, false);
        }
        // fused epilogue: tanh + dot with V, accumulate per-lane partial
        #pragma unroll
        for (int i = 0; i < 4; ++i) {
            const int n = ((ntBase + i) << 4) + lm;
            const float hv = smHP[n];
            const float vv = smV[n];
            #pragma unroll
            for (int r = 0; r < 8; ++r)
                partial[r] += tanhf(acc[i][r] + hv) * vv;
        }
    }

    // reduce partials over the 16-lane halves (C layout: M = r + 8*hi, N = lm)
    #pragma unroll
    for (int r = 0; r < 8; ++r) {
        float s = partial[r];
        s += __shfl_xor(s, 1);
        s += __shfl_xor(s, 2);
        s += __shfl_xor(s, 4);
        s += __shfl_xor(s, 8);
        if (lm == 0) {
            int row = w * 16 + hi * 8 + r;
            scores[(size_t)b * Tt + t0 + row] = s;
        }
    }
}

// ---------------------------------------------------------------------------
// Kernel M1: masked softmax over T per batch; also zero the context slice.
// ---------------------------------------------------------------------------
__global__ void __launch_bounds__(256) softmax_kernel(const float* __restrict__ scores,
                                                      const int* __restrict__ seq_len,
                                                      float* __restrict__ ctx,
                                                      float* __restrict__ attn) {
    __shared__ float red[256];
    const int b = blockIdx.x, tid = threadIdx.x;
    const int len = seq_len[b];

    float x[8]; bool ok[8];
    float m = -3.402823e38f;
    #pragma unroll
    for (int j = 0; j < 8; ++j) {
        int t = tid + j * 256;
        float s = scores[(size_t)b * Tt + t];
        ok[j] = (t < len);
        x[j] = ok[j] ? s : -1e30f;
        m = fmaxf(m, x[j]);
    }
    red[tid] = m; __syncthreads();
    for (int s = 128; s > 0; s >>= 1) {
        if (tid < s) red[tid] = fmaxf(red[tid], red[tid + s]);
        __syncthreads();
    }
    float M = red[0]; __syncthreads();

    float e[8], sum = 0.f;
    #pragma unroll
    for (int j = 0; j < 8; ++j) {
        e[j] = ok[j] ? __expf(x[j] - M) : 0.f;
        sum += e[j];
    }
    red[tid] = sum; __syncthreads();
    for (int s = 128; s > 0; s >>= 1) {
        if (tid < s) red[tid] += red[tid + s];
        __syncthreads();
    }
    float inv = 1.f / red[0];

    #pragma unroll
    for (int j = 0; j < 8; ++j)
        attn[(size_t)b * Tt + tid + j * 256] = e[j] * inv;

    #pragma unroll
    for (int j = 0; j < 4; ++j)
        ctx[(size_t)b * Dd + tid + j * 256] = 0.f;   // init for atomic accumulation
}

// ---------------------------------------------------------------------------
// Kernel M2: context[b,d] += sum_{t in chunk} attn[b,t] * features[b,t,d]
// ---------------------------------------------------------------------------
__global__ void __launch_bounds__(256) context_kernel(const float* __restrict__ features,
                                                      const float* __restrict__ attn,
                                                      float* __restrict__ ctx) {
    const int b = blockIdx.x >> 5;
    const int t0 = (blockIdx.x & 31) << 6;    // 64 t per block
    const int tid = threadIdx.x;
    float acc0 = 0.f, acc1 = 0.f, acc2 = 0.f, acc3 = 0.f;
    for (int tt = 0; tt < 64; ++tt) {
        const int t = t0 + tt;
        const float wgt = attn[(size_t)b * Tt + t];
        if (wgt == 0.f) continue;             // masked tail: skip bandwidth
        const float* fr = features + ((size_t)b * Tt + t) * Dd;
        acc0 += wgt * fr[tid];
        acc1 += wgt * fr[tid + 256];
        acc2 += wgt * fr[tid + 512];
        acc3 += wgt * fr[tid + 768];
    }
    float* c = ctx + (size_t)b * Dd;
    atomicAdd(&c[tid],       acc0);
    atomicAdd(&c[tid + 256], acc1);
    atomicAdd(&c[tid + 512], acc2);
    atomicAdd(&c[tid + 768], acc3);
}

// ---------------------------------------------------------------------------
extern "C" void kernel_launch(void* const* d_in, const int* in_sizes, int n_in,
                              void* d_out, int out_size, void* d_ws, size_t ws_size,
                              hipStream_t stream) {
    (void)in_sizes; (void)n_in; (void)out_size; (void)ws_size;
    const float* features = (const float*)d_in[0];
    const float* hidden   = (const float*)d_in[1];
    const int*   seq_len  = (const int*)d_in[2];
    const float* W1       = (const float*)d_in[3];
    const float* b1       = (const float*)d_in[4];
    const float* W2       = (const float*)d_in[5];
    const float* b2       = (const float*)d_in[6];
    const float* V        = (const float*)d_in[7];
    // d_in[8] = bV: constant shift before softmax -> cancels; unused.

    unsigned short* w1p  = (unsigned short*)d_ws;                       // 2 MB
    float* hpb    = (float*)((char*)d_ws + (2u << 20));                 // 128 KB
    float* scores = (float*)((char*)d_ws + (2u << 20) + (128u << 10));  // 256 KB

    float* ctx  = (float*)d_out;            // [32,1024]
    float* attn = ctx + Bb * Dd;            // [32,2048]

    hipFuncSetAttribute(reinterpret_cast<const void*>(score_kernel),
                        hipFuncAttributeMaxDynamicSharedMemorySize, SMEM_S);

    prep_w1       <<<4096, 256, 0, stream>>>(W1, w1p);
    hidden_proj   <<<128,  256, 0, stream>>>(hidden, W2, b1, b2, hpb);
    score_kernel  <<<(Bb * Tt) / 128, 256, SMEM_S, stream>>>(features, w1p, hpb, V, scores);
    softmax_kernel<<<Bb,   256, 0, stream>>>(scores, seq_len, ctx, attn);
    context_kernel<<<Bb * 32, 256, 0, stream>>>(features, attn, ctx);
}
